// MultiHeadSelfAttention_59974923321871
// MI455X (gfx1250) — matrix operand order repaired and checked by reading, not compiled, not run on hardware
//
#include <hip/hip_runtime.h>

typedef __attribute__((ext_vector_type(16))) __bf16 bf16x16;
typedef __attribute__((ext_vector_type(8)))  __bf16 bf16x8;
typedef __attribute__((ext_vector_type(4)))  __bf16 bf16x4;
typedef __attribute__((ext_vector_type(8)))  float  f32x8;
typedef __attribute__((ext_vector_type(4)))  float  f32x4;
typedef __attribute__((ext_vector_type(4)))  unsigned int u32x4;
typedef __attribute__((ext_vector_type(8)))  int    i32x8;
typedef __attribute__((ext_vector_type(4)))  int    i32x4;

#define B_  4
#define T_  2048
#define D_  1024
#define H_  16
#define DK_ 64

#if __has_builtin(__builtin_amdgcn_tensor_load_to_lds)
#define USE_TDM 1
#else
#define USE_TDM 0
#endif

__device__ __forceinline__ f32x8 wmma_bf16(bf16x16 a, bf16x16 b, f32x8 c) {
    return __builtin_amdgcn_wmma_f32_16x16x32_bf16(
        /*neg_a=*/false, a, /*neg_b=*/false, b,
        /*c_mod=*/(short)0, c, /*reuse_a=*/false, /*reuse_b=*/false);
}

__device__ __forceinline__ bf16x16 cat8(bf16x8 lo, bf16x8 hi) {
    return __builtin_shufflevector(lo, hi, 0,1,2,3,4,5,6,7,8,9,10,11,12,13,14,15);
}

// A operand (16x32 bf16, MxK). Lane m = lane&15 holds row m.
// Half 0 lanes hold K {0..7, 16..23}; half 1 lanes hold K {8..15, 24..31}.
__device__ __forceinline__ bf16x16 load_a_frag(const __bf16* row0, int ld, int koff, int lane) {
    const int half = lane >> 4;
    const __bf16* p = row0 + (lane & 15) * ld + koff + half * 8;
    bf16x8 lo = *(const bf16x8*)(p);
    bf16x8 hi = *(const bf16x8*)(p + 16);
    return cat8(lo, hi);
}

// B operand (32x16 bf16, KxN). Lane holds column n = lane&15; half 0 K {0..7,16..23}, half 1 K {8..15,24..31}.
__device__ __forceinline__ bf16x16 load_b_frag(const __bf16* col0, int ld, int koff, int lane) {
    const int khalf = lane >> 4;
    const __bf16* p = col0 + (lane & 15) * ld + koff + khalf * 8;
    bf16x8 lo = *(const bf16x8*)(p);
    bf16x8 hi = *(const bf16x8*)(p + 16);
    return cat8(lo, hi);
}

__device__ __forceinline__ void cvt_store4(__bf16* dst, f32x4 v) {
    dst[0] = (__bf16)v[0]; dst[1] = (__bf16)v[1];
    dst[2] = (__bf16)v[2]; dst[3] = (__bf16)v[3];
}

#if USE_TDM
// Issue a TDM 2D tile load: tile_w x tile_h bf16 elements, global row stride
// row_stride_elems, into LDS at lds_off with 8-elem (16B) per-row padding
// (64 bf16 row -> 72 bf16 LDS stride, matching LKT).
// D# layout per CDNA5 ISA 8.3/8.4 (group0 128b, group1 256b, groups 2/3 zero for 2D).
__device__ __forceinline__ void tdm_load_2d(unsigned lds_off, const void* gaddr,
                                            unsigned tile_w, unsigned tile_h,
                                            unsigned long long row_stride_elems,
                                            unsigned tensor_rows) {
    unsigned long long ga = (unsigned long long)(size_t)gaddr;
    u32x4 g0;
    g0[0] = 1u;                                            // count=1, user descriptor
    g0[1] = lds_off;                                       // lds_addr (bytes)
    g0[2] = (unsigned)(ga & 0xFFFFFFFFu);                  // global_addr[31:0]
    g0[3] = (unsigned)((ga >> 32) & 0x01FFFFFFu)           // global_addr[56:32]
          | (2u << 30);                                    // type = 2 ("image")

    const unsigned dim0 = tile_w;                          // tensor_dim0 (elems)
    const unsigned dim1 = tensor_rows;                     // tensor_dim1 (rows)
    i32x8 g1;
    g1[0] = (int)((1u << 16)        // data_size = 2 bytes
                | (1u << 20)        // pad_enable
                | (4u << 22)        // pad_interval: 32 DWORDs (= 64 bf16 row)
                | (3u << 25));      // pad_amount: 4 DWORDs (= 8 bf16 pad)
    g1[1] = (int)((dim0 & 0xFFFFu) << 16);                            // tdim0[15:0]
    g1[2] = (int)(((dim0 >> 16) & 0xFFFFu) | ((dim1 & 0xFFFFu) << 16)); // tdim0hi|tdim1lo
    g1[3] = (int)(((dim1 >> 16) & 0xFFFFu) | ((tile_w & 0xFFFFu) << 16)); // tdim1hi|tile_dim0
    g1[4] = (int)(tile_h & 0xFFFFu);                                  // tile_dim1 | tile_dim2=0
    g1[5] = (int)(unsigned)(row_stride_elems & 0xFFFFFFFFull);        // dim0_stride[31:0]
    g1[6] = (int)(unsigned)((row_stride_elems >> 32) & 0xFFFFull);    // dim0_stride[47:32]
    g1[7] = 0;                                                        // dim1_stride = 0

    i32x4 gz = {0, 0, 0, 0};
#if __clang_major__ >= 23
    i32x8 gz8 = {0, 0, 0, 0, 0, 0, 0, 0};
    __builtin_amdgcn_tensor_load_to_lds(g0, g1, gz, gz, gz8, 0);
#else
    __builtin_amdgcn_tensor_load_to_lds(g0, g1, gz, gz, 0);
#endif
}

__device__ __forceinline__ void wait_tensorcnt0() {
#if __has_builtin(__builtin_amdgcn_s_wait_tensorcnt)
    __builtin_amdgcn_s_wait_tensorcnt(0);
#else
    asm volatile("s_wait_tensorcnt 0x0" ::: "memory");
#endif
}
#endif // USE_TDM

// -------------------- Kernel 1: Q/K/V projections --------------------
// Y = X @ W^T + b ; blockIdx.z selects {Q,K,V}. Q,K stored [B,H,T,DK] bf16;
// V stored transposed [B,H,DK,T] bf16. Double-buffered LDS ping-pong.
__global__ __launch_bounds__(256) void qkv_proj_kernel(
    const float* __restrict__ q_in, const float* __restrict__ k_in, const float* __restrict__ v_in,
    const float* __restrict__ Wq, const float* __restrict__ bq,
    const float* __restrict__ Wk, const float* __restrict__ bk,
    const float* __restrict__ Wv, const float* __restrict__ bv,
    __bf16* __restrict__ qb, __bf16* __restrict__ kb, __bf16* __restrict__ vtb)
{
    constexpr int LDT = 40; // 32 + 8 pad
    __shared__ __attribute__((aligned(16))) __bf16 Xs[2][128 * LDT];
    __shared__ __attribute__((aligned(16))) __bf16 Ws[2][128 * LDT];

    const int z = blockIdx.z;
    const float* X    = (z == 0) ? q_in : (z == 1) ? k_in : v_in;
    const float* W    = (z == 0) ? Wq   : (z == 1) ? Wk   : Wv;
    const float* bias = (z == 0) ? bq   : (z == 1) ? bk   : bv;

    const int m0 = blockIdx.x * 128;   // rows of X (b*T+t)
    const int n0 = blockIdx.y * 128;   // output feature d
    const int tid = threadIdx.x;
    const int lane = tid & 31, wave = tid >> 5;

    f32x8 acc[8] = {};
    f32x4 xr[4], wr[4];

    auto load_slice = [&](int k0) {
        #pragma unroll
        for (int it = 0; it < 4; ++it) {
            int s = tid + it * 256;
            int r = s >> 3, c = (s & 7) * 4;
            xr[it] = *(const f32x4*)(X + (size_t)(m0 + r) * D_ + k0 + c);
            wr[it] = *(const f32x4*)(W + (size_t)(n0 + r) * D_ + k0 + c);
        }
    };
    auto store_slice = [&](int buf) {
        #pragma unroll
        for (int it = 0; it < 4; ++it) {
            int s = tid + it * 256;
            int r = s >> 3, c = (s & 7) * 4;
            cvt_store4(&Xs[buf][r * LDT + c], xr[it]);
            cvt_store4(&Ws[buf][r * LDT + c], wr[it]);
        }
    };

    load_slice(0);
    int buf = 0;
    for (int k0 = 0; k0 < D_; k0 += 32, buf ^= 1) {
        store_slice(buf);
        if (k0 + 32 < D_) load_slice(k0 + 32);   // global latency hides under WMMAs
        __syncthreads();
        bf16x16 af = load_a_frag(&Xs[buf][wave * 16 * LDT], LDT, 0, lane);
        bf16x16 bfr[8];
        #pragma unroll
        for (int nt = 0; nt < 8; nt++)
            bfr[nt] = load_b_frag(&Ws[buf][nt * 16 * LDT], LDT, 0, lane);
        #pragma unroll
        for (int nt = 0; nt < 8; nt++)
            acc[nt] = wmma_bf16(af, bfr[nt], acc[nt]);
        __syncthreads();
    }

    const int half = lane >> 4, n = lane & 15;
    #pragma unroll
    for (int nt = 0; nt < 8; nt++) {
        int d = n0 + nt * 16 + n;
        float bb = bias[d];
        int h = d >> 6, dk = d & (DK_ - 1);
        #pragma unroll
        for (int v = 0; v < 8; v++) {
            int row = m0 + wave * 16 + v + 8 * half;  // b*T + t
            int b = row >> 11, t = row & (T_ - 1);
            float val = acc[nt][v] + bb;
            if (z == 2) {
                vtb[(((size_t)(b * H_ + h) * DK_ + dk) * T_) + t] = (__bf16)val;
            } else {
                __bf16* dst = (z == 0) ? qb : kb;
                dst[(((size_t)(b * H_ + h) * T_ + t) * DK_) + dk] = (__bf16)val;
            }
        }
    }
}

// -------------------- Kernel 2: fused flash attention --------------------
// grid (T/128, B*H); 8 waves/block, each wave owns 16 query rows.
// K / V^T tiles staged into LDS by the Tensor Data Mover (double-buffered),
// tracked with TENSORcnt; falls back to cooperative copies if TDM builtin absent.
__global__ __launch_bounds__(256) void attn_kernel(
    const __bf16* __restrict__ qb, const __bf16* __restrict__ kb,
    const __bf16* __restrict__ vtb, const unsigned char* __restrict__ maskp,
    __bf16* __restrict__ aob)
{
    constexpr int LKT = 72; // 64 + 8 pad (matches TDM pad_interval/pad_amount)
    __shared__ __attribute__((aligned(16))) __bf16 Ks[2][64 * LKT];
    __shared__ __attribute__((aligned(16))) __bf16 Vs[2][64 * LKT];
    __shared__ __attribute__((aligned(16))) __bf16 Ps[8 * 16 * LKT];

    const int tid = threadIdx.x;
    const int lane = tid & 31, wave = tid >> 5;
    const int half = lane >> 4, n = lane & 15;
    const int bh = blockIdx.y;          // b*H + h
    const int b = bh >> 4, h = bh & 15;
    const int q0 = blockIdx.x * 128 + wave * 16;

    // Q fragments for this wave's 16 rows, kept in registers for the whole loop.
    const __bf16* qbase = qb + ((size_t)bh * T_ + q0) * DK_;
    bf16x16 qf0 = load_a_frag(qbase, DK_, 0, lane);
    bf16x16 qf1 = load_a_frag(qbase, DK_, 32, lane);

    f32x8 o[4] = {};
    float rm[8], rl[8];
    #pragma unroll
    for (int v = 0; v < 8; v++) { rm[v] = -3.0e38f; rl[v] = 0.f; }

#if USE_TDM
    auto issue_tiles = [&](int kt0, int bufn) {
        tdm_load_2d((unsigned)(size_t)&Ks[bufn][0],
                    kb + ((size_t)bh * T_ + kt0) * DK_,
                    64, 64, (unsigned long long)DK_, 64);
        tdm_load_2d((unsigned)(size_t)&Vs[bufn][0],
                    vtb + ((size_t)bh * DK_) * T_ + kt0,
                    64, 64, (unsigned long long)T_, 64);
    };
    if (wave == 0) issue_tiles(0, 0);
#else
    bf16x8 kr[2], vr[2];
    auto load_kv = [&](int kt0) {
        #pragma unroll
        for (int it = 0; it < 2; ++it) {
            int s = tid + it * 256;
            int r = s >> 3, c = (s & 7) * 8;
            kr[it] = *(const bf16x8*)(kb + ((size_t)bh * T_ + kt0 + r) * DK_ + c);
            vr[it] = *(const bf16x8*)(vtb + ((size_t)bh * DK_ + r) * T_ + kt0 + c);
        }
    };
    auto store_kv = [&](int bufn) {
        #pragma unroll
        for (int it = 0; it < 2; ++it) {
            int s = tid + it * 256;
            int r = s >> 3, c = (s & 7) * 8;
            *(bf16x8*)(&Ks[bufn][r * LKT + c]) = kr[it];
            *(bf16x8*)(&Vs[bufn][r * LKT + c]) = vr[it];
        }
    };
    load_kv(0);
#endif

    int buf = 0;
    for (int kt0 = 0; kt0 < T_; kt0 += 64, buf ^= 1) {
#if USE_TDM
        if (wave == 0) wait_tensorcnt0();     // DMA for `buf` complete
        __syncthreads();                      // publish LDS tile to all waves
        if (wave == 0 && kt0 + 64 < T_) issue_tiles(kt0 + 64, buf ^ 1); // overlap next DMA
#else
        store_kv(buf);
        if (kt0 + 64 < T_) load_kv(kt0 + 64);
        __syncthreads();
#endif

        // S = Q * K^T / sqrt(DK) + mask bias  (16 x 64)
        bf16x16 kf[8];
        #pragma unroll
        for (int nt = 0; nt < 4; nt++) {
            kf[2 * nt]     = load_b_frag(&Ks[buf][nt * 16 * LKT], LKT, 0,  lane);
            kf[2 * nt + 1] = load_b_frag(&Ks[buf][nt * 16 * LKT], LKT, 32, lane);
        }
        f32x8 S[4];
        #pragma unroll
        for (int nt = 0; nt < 4; nt++) {
            f32x8 s = {};
            s = wmma_bf16(qf0, kf[2 * nt],     s);
            s = wmma_bf16(qf1, kf[2 * nt + 1], s);
            int key = kt0 + nt * 16 + n;
            float mb = maskp[b * T_ + key] ? 0.f : -3.0e38f;
            #pragma unroll
            for (int v = 0; v < 8; v++) s[v] = s[v] * 0.125f + mb;
            S[nt] = s;
        }

        // Online softmax: rows live in VGPR index v (row = v + 8*half),
        // columns across the 16 lanes of each half -> shfl_xor reductions.
        float cmax[8];
        #pragma unroll
        for (int v = 0; v < 8; v++)
            cmax[v] = fmaxf(fmaxf(S[0][v], S[1][v]), fmaxf(S[2][v], S[3][v]));
        #pragma unroll
        for (int off = 1; off < 16; off <<= 1)
            #pragma unroll
            for (int v = 0; v < 8; v++)
                cmax[v] = fmaxf(cmax[v], __shfl_xor(cmax[v], off, 32));

        float cf[8];
        #pragma unroll
        for (int v = 0; v < 8; v++) {
            float nm = fmaxf(rm[v], cmax[v]);
            cf[v] = __expf(rm[v] - nm);
            rm[v] = nm;
            rl[v] *= cf[v];
        }
        #pragma unroll
        for (int nt = 0; nt < 4; nt++)
            #pragma unroll
            for (int v = 0; v < 8; v++) o[nt][v] *= cf[v];

        float csum[8];
        #pragma unroll
        for (int v = 0; v < 8; v++) csum[v] = 0.f;
        #pragma unroll
        for (int nt = 0; nt < 4; nt++)
            #pragma unroll
            for (int v = 0; v < 8; v++) {
                float p = __expf(S[nt][v] - rm[v]);
                S[nt][v] = p;
                csum[v] += p;
            }
        #pragma unroll
        for (int off = 1; off < 16; off <<= 1)
            #pragma unroll
            for (int v = 0; v < 8; v++) csum[v] += __shfl_xor(csum[v], off, 32);
        #pragma unroll
        for (int v = 0; v < 8; v++) rl[v] += csum[v];

        // C-layout -> A-layout conversion through per-wave LDS tile (in-order DS pipe,
        // same-wave store->load needs no barrier).
        __bf16* pw = Ps + wave * 16 * LKT;
        #pragma unroll
        for (int nt = 0; nt < 4; nt++)
            #pragma unroll
            for (int v = 0; v < 8; v++)
                pw[(v + 8 * half) * LKT + nt * 16 + n] = (__bf16)S[nt][v];

        bf16x16 pf0 = load_a_frag(pw, LKT, 0, lane);
        bf16x16 pf1 = load_a_frag(pw, LKT, 32, lane);
        bf16x16 vf[8];
        #pragma unroll
        for (int nt = 0; nt < 4; nt++) {
            vf[2 * nt]     = load_b_frag(&Vs[buf][nt * 16 * LKT], LKT, 0,  lane);
            vf[2 * nt + 1] = load_b_frag(&Vs[buf][nt * 16 * LKT], LKT, 32, lane);
        }
        #pragma unroll
        for (int nt = 0; nt < 4; nt++) {
            o[nt] = wmma_bf16(pf0, vf[2 * nt],     o[nt]);
            o[nt] = wmma_bf16(pf1, vf[2 * nt + 1], o[nt]);
        }
        __syncthreads();
    }

    // Normalize and write attn output as [B,T,D] bf16 (ready for output GEMM).
    #pragma unroll
    for (int nt = 0; nt < 4; nt++) {
        #pragma unroll
        for (int v = 0; v < 8; v++) {
            int t = q0 + v + 8 * half;
            float val = o[nt][v] / fmaxf(rl[v], 1e-20f);
            aob[((size_t)(b * T_ + t)) * D_ + h * DK_ + nt * 16 + n] = (__bf16)val;
        }
    }
}

// -------------------- Kernel 3: output projection --------------------
__global__ __launch_bounds__(256) void out_proj_kernel(
    const __bf16* __restrict__ aob, const float* __restrict__ Wo,
    const float* __restrict__ bo, float* __restrict__ out)
{
    constexpr int LDT = 40;
    __shared__ __attribute__((aligned(16))) __bf16 Xs[2][128 * LDT];
    __shared__ __attribute__((aligned(16))) __bf16 Ws[2][128 * LDT];

    const int m0 = blockIdx.x * 128, n0 = blockIdx.y * 128;
    const int tid = threadIdx.x, lane = tid & 31, wave = tid >> 5;

    f32x8 acc[8] = {};
    bf16x8 ar[2];
    f32x4  wr[4];

    auto load_slice = [&](int k0) {
        #pragma unroll
        for (int it = 0; it < 2; ++it) {
            int s = tid + it * 256;
            int r = s >> 2, c = (s & 3) * 8;
            ar[it] = *(const bf16x8*)(aob + (size_t)(m0 + r) * D_ + k0 + c);
        }
        #pragma unroll
        for (int it = 0; it < 4; ++it) {
            int s = tid + it * 256;
            int r = s >> 3, c = (s & 7) * 4;
            wr[it] = *(const f32x4*)(Wo + (size_t)(n0 + r) * D_ + k0 + c);
        }
    };
    auto store_slice = [&](int buf) {
        #pragma unroll
        for (int it = 0; it < 2; ++it) {
            int s = tid + it * 256;
            int r = s >> 2, c = (s & 3) * 8;
            *(bf16x8*)(&Xs[buf][r * LDT + c]) = ar[it];
        }
        #pragma unroll
        for (int it = 0; it < 4; ++it) {
            int s = tid + it * 256;
            int r = s >> 3, c = (s & 7) * 4;
            cvt_store4(&Ws[buf][r * LDT + c], wr[it]);
        }
    };

    load_slice(0);
    int buf = 0;
    for (int k0 = 0; k0 < D_; k0 += 32, buf ^= 1) {
        store_slice(buf);
        if (k0 + 32 < D_) load_slice(k0 + 32);
        __syncthreads();
        bf16x16 af = load_a_frag(&Xs[buf][wave * 16 * LDT], LDT, 0, lane);
        bf16x16 bfr[8];
        #pragma unroll
        for (int nt = 0; nt < 8; nt++)
            bfr[nt] = load_b_frag(&Ws[buf][nt * 16 * LDT], LDT, 0, lane);
        #pragma unroll
        for (int nt = 0; nt < 8; nt++)
            acc[nt] = wmma_bf16(af, bfr[nt], acc[nt]);
        __syncthreads();
    }

    const int half = lane >> 4, n = lane & 15;
    #pragma unroll
    for (int nt = 0; nt < 8; nt++) {
        int d = n0 + nt * 16 + n;
        float bb = bo[d];
        #pragma unroll
        for (int v = 0; v < 8; v++) {
            int row = m0 + wave * 16 + v + 8 * half;
            out[(size_t)row * D_ + d] = acc[nt][v] + bb;
        }
    }
}

extern "C" void kernel_launch(void* const* d_in, const int* in_sizes, int n_in,
                              void* d_out, int out_size, void* d_ws, size_t ws_size,
                              hipStream_t stream) {
    const float* q_in = (const float*)d_in[0];
    const float* k_in = (const float*)d_in[1];
    const float* v_in = (const float*)d_in[2];
    const unsigned char* mask = (const unsigned char*)d_in[3];
    const float* Wq = (const float*)d_in[4];
    const float* bq = (const float*)d_in[5];
    const float* Wk = (const float*)d_in[6];
    const float* bk = (const float*)d_in[7];
    const float* Wv = (const float*)d_in[8];
    const float* bv = (const float*)d_in[9];
    const float* Wo = (const float*)d_in[10];
    const float* bo = (const float*)d_in[11];
    float* out = (float*)d_out;

    const size_t elems = (size_t)B_ * H_ * T_ * DK_;  // 8,388,608
    __bf16* qb  = (__bf16*)d_ws;
    __bf16* kb  = qb + elems;
    __bf16* vtb = kb + elems;
    __bf16* aob = vtb + elems;

    // Kernel 1: Q/K/V projections (z = 0/1/2)
    {
        dim3 grid((B_ * T_) / 128, D_ / 128, 3);
        qkv_proj_kernel<<<grid, 256, 0, stream>>>(q_in, k_in, v_in,
                                                  Wq, bq, Wk, bk, Wv, bv,
                                                  qb, kb, vtb);
    }
    // Kernel 2: fused flash attention (TDM-staged tiles)
    {
        dim3 grid(T_ / 128, B_ * H_);
        attn_kernel<<<grid, 256, 0, stream>>>(qb, kb, vtb, mask, aob);
    }
    // Kernel 3: output projection
    {
        dim3 grid((B_ * T_) / 128, D_ / 128);
        out_proj_kernel<<<grid, 256, 0, stream>>>(aob, Wo, bo, out);
    }
}
